// EdgePooling_7902739824898
// MI455X (gfx1250) — matrix operand order repaired and checked by reading, not compiled, not run on hardware
//
#include <hip/hip_runtime.h>

typedef __attribute__((ext_vector_type(16))) _Float16 v16h;
typedef __attribute__((ext_vector_type(8)))  float    v8f;

#define DN 128
#define DE 32
#define ADD_SCORE 0.5f
#define TIE_EPS 1e-9f

__device__ __forceinline__ void atomicMaxF(float* addr, float v) {
  // works for any float when the slot is initialized to -inf
  if (v >= 0.0f) atomicMax((int*)addr, __float_as_int(v));
  else           atomicMin((unsigned int*)addr, (unsigned int)__float_as_int(v));
}

// ---------------- init / zero ----------------
__global__ void k_zero(float* p, int n) {
  int i = blockIdx.x * blockDim.x + threadIdx.x;
  if (i < n) p[i] = 0.0f;
}

__global__ void k_init_nodes(float* m, float* den, float* best, int* rep,
                             float* cscore, int N) {
  int i = blockIdx.x * blockDim.x + threadIdx.x;
  if (i <= N) rep[i] = i;
  if (i < N) {
    m[i]    = -__builtin_inff();
    den[i]  = 0.0f;
    best[i] = -__builtin_inff();
    cscore[i] = 1.0f;
  }
}

// ---------------- WMMA GEMVs ----------------
// s_src[i] = node_feat[i]·w_src ; s_dst[i] = node_feat[i]·w_dst
// One wave per 16 rows. B columns: 0 = w_src chunk, 1 = w_dst chunk, rest 0.
__global__ void k_node_gemv(const float* __restrict__ nf,
                            const float* __restrict__ wsrc,
                            const float* __restrict__ wdst,
                            float* __restrict__ sS, float* __restrict__ sD,
                            int ntiles) {
  int wave = (int)((blockIdx.x * blockDim.x + threadIdx.x) >> 5);
  int lane = threadIdx.x & 31;
  if (wave >= ntiles) return;              // uniform per wave -> EXEC all-ones
  const int rows = wave * 16;
  const int mrow = lane & 15;
  const int kbA  = (lane >> 4) * 16;       // A: lanes 16-31 hold K=16..31
  const int ncol = lane & 15;              // B: column index
  const int kbB  = (lane >> 4) * 16;       // B: lanes 16-31 hold K=16..31
  const float* arow = nf + (size_t)(rows + mrow) * DN;

  v8f c = {};
  for (int k0 = 0; k0 < DN; k0 += 32) {
    v16h a, b;
#pragma unroll
    for (int j = 0; j < 8; ++j) {
      a[j]     = (_Float16)arow[k0 + kbA + j];
      a[j + 8] = (_Float16)arow[k0 + 8 + kbA + j];
    }
#pragma unroll
    for (int j = 0; j < 16; ++j) b[j] = (_Float16)0.0f;
    if (ncol == 0) {
#pragma unroll
      for (int j = 0; j < 16; ++j) b[j] = (_Float16)wsrc[k0 + kbB + j];
    } else if (ncol == 1) {
#pragma unroll
      for (int j = 0; j < 16; ++j) b[j] = (_Float16)wdst[k0 + kbB + j];
    }
    c = __builtin_amdgcn_wmma_f32_16x16x32_f16(false, a, false, b,
                                               (short)0, c, false, false);
  }
  // D layout: VGPR r, lanes 0-15 -> M=r ; lanes 16-31 -> M=8+r ; N = lane&15
  int base = rows + (lane >> 4) * 8;
  if (ncol == 0) {
#pragma unroll
    for (int r = 0; r < 8; ++r) sS[base + r] = c[r];
  } else if (ncol == 1) {
#pragma unroll
    for (int r = 0; r < 8; ++r) sD[base + r] = c[r];
  }
}

// dot[e] = edge_feat[e]·w_edge  (K = 32 exactly -> one WMMA per 16 edges)
// The 16x32 f32 tile (2KB) is staged into LDS with async-to-LDS copies
// (ASYNCcnt path), then the WMMA A-fragment is built from LDS.
__global__ void k_edge_gemv(const float* __restrict__ ef,
                            const float* __restrict__ wedge,
                            float* __restrict__ dotv, int ntiles) {
  __shared__ float smem[8 * 16 * DE];      // 8 waves/block * 2KB
  int wave = (int)((blockIdx.x * blockDim.x + threadIdx.x) >> 5);
  int lane = threadIdx.x & 31;
  int wib  = threadIdx.x >> 5;
  if (wave >= ntiles) return;
  const int rows = wave * 16;
  float* wbuf = &smem[wib * 16 * DE];

  // ---- async stage: 4 x b128 per lane = 2048 bytes per wave ----
  {
    const char* gbase = (const char*)(ef + (size_t)rows * DE);
    __builtin_prefetch(gbase + 2048, 0, 1);              // next tile stream hint
    unsigned lbase = (unsigned)(size_t)wbuf;
#pragma unroll
    for (int i = 0; i < 4; ++i) {
      unsigned           loff  = lbase + (unsigned)(i * 512 + lane * 16);
      unsigned long long gaddr = (unsigned long long)(gbase + i * 512 + lane * 16);
      asm volatile("global_load_async_to_lds_b128 %0, %1, off"
                   :: "v"(loff), "v"(gaddr) : "memory");
    }
    asm volatile("s_wait_asynccnt 0x0" ::: "memory");
  }

  const int mrow = lane & 15;
  const int kbA  = (lane >> 4) * 16;
  const int ncol = lane & 15;
  const int kbB  = (lane >> 4) * 16;
  const float* arow = wbuf + mrow * DE;

  v16h a, b;
#pragma unroll
  for (int j = 0; j < 8; ++j) {
    a[j]     = (_Float16)arow[kbA + j];
    a[j + 8] = (_Float16)arow[8 + kbA + j];
  }
#pragma unroll
  for (int j = 0; j < 16; ++j) b[j] = (_Float16)0.0f;
  if (ncol == 0) {
#pragma unroll
    for (int j = 0; j < 16; ++j) b[j] = (_Float16)wedge[kbB + j];
  }
  v8f c = {};
  c = __builtin_amdgcn_wmma_f32_16x16x32_f16(false, a, false, b,
                                             (short)0, c, false, false);
  if (ncol == 0) {
    int base = rows + (lane >> 4) * 8;
#pragma unroll
    for (int r = 0; r < 8; ++r) dotv[base + r] = c[r];
  }
}

// ---------------- softmax over dst ----------------
__global__ void k_eraw_max(const float* sS, const float* sD,
                           const int* src, const int* dst,
                           float* ev, float* m, int E) {
  int i = blockIdx.x * blockDim.x + threadIdx.x;
  if (i >= E) return;
  float v = sS[src[i]] + sD[dst[i]] + ev[i];
  ev[i] = v;
  atomicMaxF(&m[dst[i]], v);
}

__global__ void k_exp_sum(float* ev, const float* m, const int* dst,
                          float* den, int E) {
  int i = blockIdx.x * blockDim.x + threadIdx.x;
  if (i >= E) return;
  float x = __expf(ev[i] - m[dst[i]]);
  ev[i] = x;
  atomicAdd(&den[dst[i]], x);
}

__global__ void k_norm_best(const float* ev, const float* den,
                            const int* src, const int* dst,
                            float* e_out, float* eu, float* best, int E) {
  int i = blockIdx.x * blockDim.x + threadIdx.x;
  if (i >= E) return;
  float e = ev[i] / den[dst[i]] + ADD_SCORE;
  e_out[i] = e;
  float u = e + (float)i * TIE_EPS;
  eu[i] = u;
  atomicMaxF(&best[src[i]], u);
  atomicMaxF(&best[dst[i]], u);
}

// ---------------- mutual matching + rep scatter-min ----------------
__global__ void k_mutual_rep(const float* eu, const float* best,
                             const int* src, const int* dst,
                             int* rep, int E) {
  int i = blockIdx.x * blockDim.x + threadIdx.x;
  if (i >= E) return;
  int s = src[i], d = dst[i];
  float u = eu[i];
  bool mut = (u >= best[s]) && (u >= best[d]) && (s != d);
  if (mut) {
    int mn = s < d ? s : d;
    atomicMin(&rep[s], mn);
    atomicMin(&rep[d], mn);
  }
}

// ---------------- single-workgroup scans ----------------
__global__ void k_newid_scan(const int* rep, int* newid, int* cluster, int N) {
  __shared__ int tmp[1024];
  __shared__ int carry;
  int tid = threadIdx.x;
  if (tid == 0) carry = 0;
  __syncthreads();
  for (int base = 0; base < N; base += 1024) {
    int i = base + tid;
    int v = (i < N && rep[i] == i) ? 1 : 0;
    tmp[tid] = v;
    __syncthreads();
    for (int off = 1; off < 1024; off <<= 1) {
      int t = (tid >= off) ? tmp[tid - off] : 0;
      __syncthreads();
      tmp[tid] += t;
      __syncthreads();
    }
    if (i < N) newid[i] = carry + tmp[tid] - 1;
    int total = tmp[1023];
    __syncthreads();
    if (tid == 0) carry += total;
    __syncthreads();
  }
  for (int i = tid; i < N; i += 1024) cluster[i] = newid[rep[i]];
}

__global__ void k_seg_scan(const unsigned int* ks, int* seg, int* nuniq, int E) {
  __shared__ int tmp[1024];
  __shared__ int carry;
  int tid = threadIdx.x;
  if (tid == 0) carry = 0;
  __syncthreads();
  for (int base = 0; base < E; base += 1024) {
    int i = base + tid;
    int v = 0;
    if (i < E) v = (i == 0) ? 1 : (ks[i] != ks[i - 1] ? 1 : 0);
    tmp[tid] = v;
    __syncthreads();
    for (int off = 1; off < 1024; off <<= 1) {
      int t = (tid >= off) ? tmp[tid - off] : 0;
      __syncthreads();
      tmp[tid] += t;
      __syncthreads();
    }
    if (i < E) seg[i] = carry + tmp[tid] - 1;
    int total = tmp[1023];
    __syncthreads();
    if (tid == 0) carry += total;
    __syncthreads();
  }
  if (tid == 0) nuniq[0] = seg[E - 1] + 1;
}

// ---------------- clustering outputs ----------------
__global__ void k_cscore(const float* eu, const float* best, const float* e_out,
                         const int* src, const int* dst, const int* cluster,
                         float* cscore, int E) {
  int i = blockIdx.x * blockDim.x + threadIdx.x;
  if (i >= E) return;
  int s = src[i], d = dst[i];
  float u = eu[i];
  bool mut = (u >= best[s]) && (u >= best[d]) && (s != d);
  if (mut) cscore[cluster[s]] = e_out[i];   // duplicate winners race (JAX scatter-set semantics)
}

__global__ void k_make_keys(const int* src, const int* dst, const int* cluster,
                            unsigned int* key, int* ord, int N, int E) {
  int i = blockIdx.x * blockDim.x + threadIdx.x;
  if (i >= E) return;
  key[i] = (unsigned int)cluster[src[i]] * (unsigned int)N +
           (unsigned int)cluster[dst[i]];
  ord[i] = i;
}

__global__ void k_out_cluster(const int* cluster, float* o_cluster, int N) {
  int i = blockIdx.x * blockDim.x + threadIdx.x;
  if (i < N) o_cluster[i] = (float)cluster[i];
}

// ---------------- radix sort (4 x 8-bit LSD, stable) ----------------
__global__ void k_hist_zero(int* hist) { hist[threadIdx.x] = 0; }

__global__ void k_hist(const unsigned int* key, int* hist, int E, int shift) {
  int stride = gridDim.x * blockDim.x;
  for (int i = blockIdx.x * blockDim.x + threadIdx.x; i < E; i += stride)
    atomicAdd(&hist[(key[i] >> shift) & 255u], 1);
}

__global__ void k_hist_scan(int* hist) {
  int sum = 0;
  for (int d = 0; d < 256; ++d) { int c = hist[d]; hist[d] = sum; sum += c; }
}

__global__ void k_scatter(const unsigned int* kin, const int* oin,
                          unsigned int* kout, int* oout, int* hist,
                          int E, int shift) {
  for (int i = 0; i < E; ++i) {            // single thread -> stable
    unsigned int d = (kin[i] >> shift) & 255u;
    int p = hist[d]++;
    kout[p] = kin[i];
    oout[p] = oin[i];
  }
}

// ---------------- dedup aggregation + final outputs ----------------
__global__ void k_ukey(const unsigned int* ks, const int* seg,
                       unsigned int* ukey, int E) {
  int i = blockIdx.x * blockDim.x + threadIdx.x;
  if (i < E) ukey[seg[i]] = ks[i];
}

__global__ void k_agg(const float* __restrict__ ef, const int* seg,
                      const int* ord, float* nef, int E) {
  int i = blockIdx.x * blockDim.x + threadIdx.x;
  if (i >= E) return;
  int sgi = seg[i];
  const float* row = ef + (size_t)ord[i] * DE;
  float* out = nef + (size_t)sgi * DE;
#pragma unroll
  for (int d = 0; d < DE; ++d) atomicAdd(&out[d], row[d]);
}

__global__ void k_final(const unsigned int* ukey, const int* nuniq,
                        float* o_nsrc, float* o_ndst, float* o_valid,
                        float* nef, int N, int E) {
  int i = blockIdx.x * blockDim.x + threadIdx.x;
  if (i >= E) return;
  int nu = nuniq[0];
  unsigned int k = ukey[i];
  unsigned int us = k / (unsigned int)N;
  unsigned int ud = k % (unsigned int)N;
  bool valid = (i < nu) && (us != ud);
  o_nsrc[i]  = valid ? (float)us : -1.0f;
  o_ndst[i]  = valid ? (float)ud : -1.0f;
  o_valid[i] = valid ? 1.0f : 0.0f;
  if (!valid && i < nu) {
    float* row = nef + (size_t)i * DE;
#pragma unroll
    for (int d = 0; d < DE; ++d) row[d] = 0.0f;
  }
}

// ---------------- launch ----------------
extern "C" void kernel_launch(void* const* d_in, const int* in_sizes, int n_in,
                              void* d_out, int out_size, void* d_ws, size_t ws_size,
                              hipStream_t stream) {
  const float* node_feat = (const float*)d_in[0];
  const float* edge_feat = (const float*)d_in[1];
  const int*   src       = (const int*)d_in[2];
  const int*   dst       = (const int*)d_in[3];
  const float* w_src     = (const float*)d_in[4];
  const float* w_dst     = (const float*)d_in[5];
  const float* w_edge    = (const float*)d_in[6];

  const int N = in_sizes[0] / DN;
  const int E = in_sizes[2];

  // output sections (all flattened to float in return order)
  float* out       = (float*)d_out;
  float* o_nsrc    = out;
  float* o_ndst    = out + (size_t)E;
  float* o_valid   = out + 2 * (size_t)E;
  float* o_cluster = out + 3 * (size_t)E;
  float* o_cscore  = out + 3 * (size_t)E + (size_t)N;
  float* o_nef     = out + 3 * (size_t)E + 2 * (size_t)N;
  float* o_e       = out + 35 * (size_t)E + 2 * (size_t)N;

  // workspace carve-up (256B aligned slices)
  char* ws = (char*)d_ws;
  size_t off = 0;
  auto take = [&](size_t bytes) {
    off = (off + 255) & ~(size_t)255;
    size_t o = off; off += bytes; return (void*)(ws + o);
  };
  float* sS    = (float*)take((size_t)N * 4);
  float* sD    = (float*)take((size_t)N * 4);
  float* m_    = (float*)take((size_t)N * 4);
  float* den   = (float*)take((size_t)N * 4);
  float* best  = (float*)take((size_t)N * 4);
  int*   rep   = (int*)take(((size_t)N + 1) * 4);
  int*   newid = (int*)take((size_t)N * 4);
  int*   clus  = (int*)take((size_t)N * 4);
  int*   hist  = (int*)take(256 * 4);
  int*   nuniq = (int*)take(4);
  float* ev    = (float*)take((size_t)E * 4);   // dot -> e_raw -> ex ; later reused as seg
  float* eu    = (float*)take((size_t)E * 4);
  unsigned int* keyA = (unsigned int*)take((size_t)E * 4);
  unsigned int* keyB = (unsigned int*)take((size_t)E * 4); // later reused as ukey
  int* ordA  = (int*)take((size_t)E * 4);
  int* ordB  = (int*)take((size_t)E * 4);
  int* seg   = (int*)ev;   // alias: ev dead after softmax normalization

  const int TB = 256;
  const int gE = (E + TB - 1) / TB;
  const int gN1 = (N + 1 + TB - 1) / TB;
  const int gOut = (out_size + TB - 1) / TB;

  // 0) zero outputs, init node-side state
  k_zero<<<gOut, TB, 0, stream>>>(out, out_size);
  k_init_nodes<<<gN1, TB, 0, stream>>>(m_, den, best, rep, o_cscore, N);

  // 1) WMMA GEMVs
  {
    int ntiles = N / 16;                         // 2500
    int blocks = (ntiles * 32 + TB - 1) / TB;
    k_node_gemv<<<blocks, TB, 0, stream>>>(node_feat, w_src, w_dst, sS, sD, ntiles);
  }
  {
    int ntiles = E / 16;                         // 80000
    int blocks = (ntiles * 32 + TB - 1) / TB;
    k_edge_gemv<<<blocks, TB, 0, stream>>>(edge_feat, w_edge, ev, ntiles);
  }

  // 2) softmax over dst
  k_eraw_max<<<gE, TB, 0, stream>>>(sS, sD, src, dst, ev, m_, E);
  k_exp_sum<<<gE, TB, 0, stream>>>(ev, m_, dst, den, E);
  k_norm_best<<<gE, TB, 0, stream>>>(ev, den, src, dst, o_e, eu, best, E);

  // 3) mutual matching + representative scatter-min
  k_mutual_rep<<<gE, TB, 0, stream>>>(eu, best, src, dst, rep, E);

  // 4) relabel clusters (prefix sum) and cluster scores
  k_newid_scan<<<1, 1024, 0, stream>>>(rep, newid, clus, N);
  k_cscore<<<gE, TB, 0, stream>>>(eu, best, o_e, src, dst, clus, o_cscore, E);
  k_out_cluster<<<(N + TB - 1) / TB, TB, 0, stream>>>(clus, o_cluster, N);

  // 5) keys + radix sort (stable LSD, 4 passes, ping-pong A->B->A->B->A)
  k_make_keys<<<gE, TB, 0, stream>>>(src, dst, clus, keyA, ordA, N, E);
  unsigned int* kin = keyA; int* oin = ordA;
  unsigned int* kout = keyB; int* oout = ordB;
  for (int pass = 0; pass < 4; ++pass) {
    int shift = pass * 8;
    k_hist_zero<<<1, 256, 0, stream>>>(hist);
    k_hist<<<256, 256, 0, stream>>>(kin, hist, E, shift);
    k_hist_scan<<<1, 1, 0, stream>>>(hist);
    k_scatter<<<1, 1, 0, stream>>>(kin, oin, kout, oout, hist, E, shift);
    unsigned int* tk = kin; kin = kout; kout = tk;
    int* to = oin; oin = oout; oout = to;
  }
  // sorted keys in kin(==keyA), permutation in oin(==ordA)

  // 6) segment ids over sorted keys, unique keys, aggregation
  k_seg_scan<<<1, 1024, 0, stream>>>(kin, seg, nuniq, E);
  k_ukey<<<gE, TB, 0, stream>>>(kin, seg, keyB, E);   // keyB reused as ukey
  k_agg<<<gE, TB, 0, stream>>>(edge_feat, seg, oin, o_nef, E);

  // 7) final edge outputs + masking of self-loop clusters
  k_final<<<gE, TB, 0, stream>>>(keyB, nuniq, o_nsrc, o_ndst, o_valid, o_nef, N, E);
}
